// TRMGATLayer_36532991820557
// MI455X (gfx1250) — compile-verified, matrix operand order checked by reading, not run on hardware
//
#include <hip/hip_runtime.h>
#include <cstdint>

// TRMGAT layer for gfx1250 (MI455X): f16 WMMA GEMMs + flash-attention with
// online softmax. All matmuls go through v_wmma_f32_16x16x32_f16 (wave32).
// K tiles staged via global_load_async_to_lds_b128; softmax row reductions
// via v_permlane16_b32 (VALU) instead of ds_bpermute.

typedef __attribute__((ext_vector_type(16))) _Float16 v16h;
typedef __attribute__((ext_vector_type(8)))  _Float16 v8h;
typedef __attribute__((ext_vector_type(8)))  float    v8f;

#define N_NODES 4096
#define DMODEL  128
#define QKD     32
#define NHEAD   8
#define HID     (DMODEL * NHEAD)   // 1024
#define QKH     (QKD * NHEAD)      // 256
#define LDSPAD  40                 // halves per row: 80B stride, 16B aligned

#ifndef __has_builtin
#define __has_builtin(x) 0
#endif

#if __has_builtin(__builtin_amdgcn_global_load_async_to_lds_b128) && \
    __has_builtin(__builtin_amdgcn_s_wait_asynccnt)
#define HAVE_ASYNC_LDS 1
#else
#define HAVE_ASYNC_LDS 0
#endif

#if HAVE_ASYNC_LDS
// Builtin expects pointers to 16-byte int vectors (per hipcc diagnostic:
// "__attribute__((__vector_size__(4 * sizeof(int)))) int __device__ *").
// Build addrspace-qualified pointers via integer round-trip (LDS offset lives
// in the low 32 bits of a generic shared address per the aperture mapping).
typedef int vs4i __attribute__((vector_size(16)));
#define AS1CAST(p) ((__attribute__((address_space(1))) vs4i*)(unsigned long long)(p))
#define AS3CAST(p) ((__attribute__((address_space(3))) vs4i*)(unsigned int)(unsigned long long)(p))
#endif

static __device__ __forceinline__ v16h cat16(v8h lo, v8h hi) {
  return __builtin_shufflevector(lo, hi, 0, 1, 2, 3, 4, 5, 6, 7,
                                 8, 9, 10, 11, 12, 13, 14, 15);
}

static __device__ __forceinline__ v8f wmma16(v16h a, v16h b, v8f c) {
  // D = A(16x32 f16) * B(32x16 f16) + C(16x16 f32)
  return __builtin_amdgcn_wmma_f32_16x16x32_f16(false, a, false, b,
                                                (short)0, c, false, false);
}

// Load a 16-bit A/B fragment: per-lane row pointer `p` holds 32 contiguous
// K-halves. ISA K-map: lanes 0-15 -> K[0..7],K[16..23]; lanes 16-31 ->
// K[8..15],K[24..31].
static __device__ __forceinline__ v16h frag_ld(const _Float16* p, int lane) {
  const int kb = (lane >> 4) << 3;
  v8h lo = *(const v8h*)(p + kb);
  v8h hi = *(const v8h*)(p + 16 + kb);
  return cat16(lo, hi);
}

#if __has_builtin(__builtin_amdgcn_permlane16)
// xor-lane exchange inside each 16-lane row via v_permlane16_b32 (VALU only,
// co-executes with WMMA; avoids ds_bpermute traffic in the hot loop).
static __device__ __forceinline__ float lanex(float v, unsigned lo, unsigned hi) {
  unsigned u = __float_as_uint(v);
  u = __builtin_amdgcn_permlane16(u, u, lo, hi, false, false);
  return __uint_as_float(u);
}
static __device__ __forceinline__ float rmax16(float v) {
  v = fmaxf(v, lanex(v, 0x67452301u, 0xEFCDAB89u));  // xor 1
  v = fmaxf(v, lanex(v, 0x54761032u, 0xDCFE98BAu));  // xor 2
  v = fmaxf(v, lanex(v, 0x32107654u, 0xBA98FEDCu));  // xor 4
  v = fmaxf(v, lanex(v, 0xFEDCBA98u, 0x76543210u));  // xor 8
  return v;
}
static __device__ __forceinline__ float rsum16(float v) {
  v += lanex(v, 0x67452301u, 0xEFCDAB89u);
  v += lanex(v, 0x54761032u, 0xDCFE98BAu);
  v += lanex(v, 0x32107654u, 0xBA98FEDCu);
  v += lanex(v, 0xFEDCBA98u, 0x76543210u);
  return v;
}
#else
static __device__ __forceinline__ float rmax16(float v) {
  v = fmaxf(v, __shfl_xor(v, 1, 32));
  v = fmaxf(v, __shfl_xor(v, 2, 32));
  v = fmaxf(v, __shfl_xor(v, 4, 32));
  v = fmaxf(v, __shfl_xor(v, 8, 32));
  return v;
}
static __device__ __forceinline__ float rsum16(float v) {
  v += __shfl_xor(v, 1, 32);
  v += __shfl_xor(v, 2, 32);
  v += __shfl_xor(v, 4, 32);
  v += __shfl_xor(v, 8, 32);
  return v;
}
#endif

__global__ void cvt_f32_f16(const float* __restrict__ in,
                            _Float16* __restrict__ out, int n) {
  int i = blockIdx.x * blockDim.x + threadIdx.x;
  if (i < n) out[i] = (_Float16)in[i];
}

// C[M,N] = A[M,K](f16) * B[K,N](f16) + bias.  Block: 256 thr = 8 waves,
// tile 128 rows x 64 cols (each wave: 16 rows x 64 cols).
template <bool HALF_OUT>
__global__ __launch_bounds__(256) void gemm_wmma(
    const _Float16* __restrict__ A, const _Float16* __restrict__ B,
    const float* __restrict__ bias, void* __restrict__ outp, int K, int Nn) {
  __shared__ _Float16 Bt[64][LDSPAD];  // B tile transposed: [col][k]

  const int lane = threadIdx.x & 31;
  const int w = threadIdx.x >> 5;
  const int rowbase = blockIdx.y * 128 + w * 16;
  const int cbase = blockIdx.x * 64;
  const int hi = lane >> 4;
  const int c = lane & 15;

  v8f acc[4] = {};
  const _Float16* arow = A + (size_t)(rowbase + c) * K;

  for (int kt = 0; kt < K; kt += 32) {
    __syncthreads();
    {  // cooperative transpose-stage of B[kt..kt+32)[cbase..cbase+64)
      const int k = threadIdx.x & 31;
      const int cq = threadIdx.x >> 5;  // 0..7
      const _Float16* brow = B + (size_t)(kt + k) * Nn + cbase + cq * 8;
      v8h bv = *(const v8h*)brow;
#pragma unroll
      for (int j = 0; j < 8; ++j) Bt[cq * 8 + j][k] = bv[j];
    }
    __syncthreads();

    v16h af = frag_ld(arow + kt, lane);
#pragma unroll
    for (int cg = 0; cg < 4; ++cg) {
      v16h bf = frag_ld(&Bt[cg * 16 + c][0], lane);
      acc[cg] = wmma16(af, bf, acc[cg]);
    }
  }

#pragma unroll
  for (int cg = 0; cg < 4; ++cg) {
#pragma unroll
    for (int r = 0; r < 8; ++r) {
      const int row = rowbase + r + 8 * hi;
      const int col = cbase + cg * 16 + c;
      const float v = acc[cg][r] + bias[col];
      if (HALF_OUT)
        ((_Float16*)outp)[(size_t)row * Nn + col] = (_Float16)v;
      else
        ((float*)outp)[(size_t)row * Nn + col] = v;
    }
  }
}

// Flash attention: grid (N/128, NHEAD), 256 thr. Each wave: 16 q-rows,
// f32 16x128 accumulator, online softmax over 32-key tiles.
__global__ __launch_bounds__(256) void attn_wmma(
    const _Float16* __restrict__ qh, const _Float16* __restrict__ kh,
    const _Float16* __restrict__ vh, const float* __restrict__ edge_bias,
    const unsigned char* __restrict__ mask, _Float16* __restrict__ resh) {
  __shared__ _Float16 Vt[DMODEL][LDSPAD];  // V tile transposed: [ch][key]
  __shared__ _Float16 Kst[32][QKD];        // K tile row-major: [key][qk-dim]
  __shared__ _Float16 Pst[8][16][LDSPAD];  // per-wave P staging (C->A relayout)

  const int lane = threadIdx.x & 31;
  const int w = threadIdx.x >> 5;
  const int h = blockIdx.y;
  const int qbase = blockIdx.x * 128 + w * 16;
  const int hi = lane >> 4;
  const int c = lane & 15;

  const v16h qf = frag_ld(qh + (size_t)(qbase + c) * QKH + h * QKD, lane);

  v8f acc[8] = {};
  float mrow[8], lrow[8];
#pragma unroll
  for (int r = 0; r < 8; ++r) { mrow[r] = -1e30f; lrow[r] = 0.f; }
  const float isq = 0.17677669529663687f;  // 1/sqrt(32)

  for (int kt = 0; kt < N_NODES; kt += 32) {
    __syncthreads();
    {
      // --- K tile: async DMA to LDS (wave 0), overlapped with V staging ---
#if HAVE_ASYNC_LDS
      if (w == 0) {
#pragma unroll
        for (int i = 0; i < 4; ++i) {
          const int row = i * 8 + (lane >> 2);
          const int seg = (lane & 3) * 8;  // halves
          const _Float16* g = kh + (size_t)(kt + row) * QKH + h * QKD + seg;
          __builtin_amdgcn_global_load_async_to_lds_b128(
              AS1CAST(g), AS3CAST(&Kst[row][seg]), 0, 0);
        }
      }
#else
      if (threadIdx.x < 128) {
        const int row = threadIdx.x >> 2;
        const int seg = (threadIdx.x & 3) * 8;
        *(v8h*)&Kst[row][seg] =
            *(const v8h*)(kh + (size_t)(kt + row) * QKH + h * QKD + seg);
      }
#endif
      // --- V tile: stage transposed into LDS (all 256 threads) ---
      const int klocal = threadIdx.x >> 3;      // 0..31
      const int cseg = (threadIdx.x & 7) * 16;  // 0..112
      const _Float16* vrow = vh + (size_t)(kt + klocal) * HID + h * DMODEL + cseg;
      v8h a0 = *(const v8h*)vrow;
      v8h a1 = *(const v8h*)(vrow + 8);
#pragma unroll
      for (int j = 0; j < 8; ++j) {
        Vt[cseg + j][klocal] = a0[j];
        Vt[cseg + 8 + j][klocal] = a1[j];
      }
      if (kt + 32 < N_NODES)  // global_prefetch of next V tile
        __builtin_prefetch(vh + (size_t)(kt + 32 + klocal) * HID + h * DMODEL + cseg, 0, 1);
#if HAVE_ASYNC_LDS
      if (w == 0) __builtin_amdgcn_s_wait_asynccnt(0);  // drain before barrier
#endif
    }
    __syncthreads();

    // S = Q K^T over this 32-key tile (two 16x16 WMMA tiles), K from LDS
    v16h kf0 = frag_ld(&Kst[c][0], lane);
    v16h kf1 = frag_ld(&Kst[16 + c][0], lane);
    v8f z = {};
    v8f s0 = wmma16(qf, kf0, z);
    v8f s1 = wmma16(qf, kf1, z);

#pragma unroll
    for (int r = 0; r < 8; ++r) {
      const int row = qbase + r + 8 * hi;
      const size_t rb = (size_t)row * N_NODES;
      const int col0 = kt + c;
      const int col1 = kt + 16 + c;
      float a = s0[r] * isq;
      float b = s1[r] * isq;
      if (h == 0) {
        a += edge_bias[rb + col0];
        b += edge_bias[rb + col1];
      }
      if (mask[rb + col0]) a = -1e9f;
      if (mask[rb + col1]) b = -1e9f;

      const float tm = rmax16(fmaxf(a, b));
      const float mnew = fmaxf(mrow[r], tm);
      const float sc = __expf(mrow[r] - mnew);
      const float p0 = __expf(a - mnew);
      const float p1 = __expf(b - mnew);
      lrow[r] = lrow[r] * sc + rsum16(p0 + p1);
      mrow[r] = mnew;
#pragma unroll
      for (int cg = 0; cg < 8; ++cg) acc[cg][r] *= sc;

      const int ml = r + 8 * hi;  // C-layout row -> LDS row
      Pst[w][ml][c] = (_Float16)p0;
      Pst[w][ml][16 + c] = (_Float16)p1;
    }

    // P (A-frag via LDS relayout) x V (B-frag from transposed tile)
    v16h pf = frag_ld(&Pst[w][c][0], lane);
#pragma unroll
    for (int cg = 0; cg < 8; ++cg) {
      v16h vf = frag_ld(&Vt[cg * 16 + c][0], lane);
      acc[cg] = wmma16(pf, vf, acc[cg]);
    }
  }

#pragma unroll
  for (int r = 0; r < 8; ++r) {
    const int row = qbase + r + 8 * hi;
    const float inv = 1.0f / lrow[r];
#pragma unroll
    for (int cg = 0; cg < 8; ++cg) {
      const int col = h * DMODEL + cg * 16 + c;
      resh[(size_t)row * HID + col] = (_Float16)(acc[cg][r] * inv);
    }
  }
}

extern "C" void kernel_launch(void* const* d_in, const int* in_sizes, int n_in,
                              void* d_out, int out_size, void* d_ws, size_t ws_size,
                              hipStream_t stream) {
  (void)in_sizes; (void)n_in; (void)out_size; (void)ws_size;
  const float* x  = (const float*)d_in[0];
  const unsigned char* mask = (const unsigned char*)d_in[1];  // jnp bool_: 1 byte
  const float* eb = (const float*)d_in[2];
  const float* Wq = (const float*)d_in[3];
  const float* bq = (const float*)d_in[4];
  const float* Wk = (const float*)d_in[5];
  const float* bk = (const float*)d_in[6];
  const float* Wv = (const float*)d_in[7];
  const float* bv = (const float*)d_in[8];
  const float* Wo = (const float*)d_in[9];
  const float* bo = (const float*)d_in[10];

  char* ws = (char*)d_ws;  // ~21.7 MB total
  _Float16* xh   = (_Float16*)(ws + 0);         // 1,048,576 B
  _Float16* wqh  = (_Float16*)(ws + 1048576);   //    65,536 B
  _Float16* wkh  = (_Float16*)(ws + 1114112);   //    65,536 B
  _Float16* wvh  = (_Float16*)(ws + 1179648);   //   262,144 B
  _Float16* woh  = (_Float16*)(ws + 1441792);   //   262,144 B
  _Float16* qhp  = (_Float16*)(ws + 1703936);   // 2,097,152 B
  _Float16* khp  = (_Float16*)(ws + 3801088);   // 2,097,152 B
  _Float16* vhp  = (_Float16*)(ws + 5898240);   // 8,388,608 B
  _Float16* resh = (_Float16*)(ws + 14286848);  // 8,388,608 B

  cvt_f32_f16<<<(N_NODES * DMODEL + 255) / 256, 256, 0, stream>>>(x, xh, N_NODES * DMODEL);
  cvt_f32_f16<<<(DMODEL * QKH + 255) / 256, 256, 0, stream>>>(Wq, wqh, DMODEL * QKH);
  cvt_f32_f16<<<(DMODEL * QKH + 255) / 256, 256, 0, stream>>>(Wk, wkh, DMODEL * QKH);
  cvt_f32_f16<<<(DMODEL * HID + 255) / 256, 256, 0, stream>>>(Wv, wvh, DMODEL * HID);
  cvt_f32_f16<<<(HID * DMODEL + 255) / 256, 256, 0, stream>>>(Wo, woh, HID * DMODEL);

  // Projections: q,k [4096,256], v [4096,1024] (f16 outputs for attention)
  gemm_wmma<true><<<dim3(QKH / 64, N_NODES / 128), 256, 0, stream>>>(xh, wqh, bq, qhp, DMODEL, QKH);
  gemm_wmma<true><<<dim3(QKH / 64, N_NODES / 128), 256, 0, stream>>>(xh, wkh, bk, khp, DMODEL, QKH);
  gemm_wmma<true><<<dim3(HID / 64, N_NODES / 128), 256, 0, stream>>>(xh, wvh, bv, vhp, DMODEL, HID);

  // Flash attention over all heads
  attn_wmma<<<dim3(N_NODES / 128, NHEAD), 256, 0, stream>>>(qhp, khp, vhp, eb, mask, resh);

  // Output projection -> f32 d_out [4096,128]
  gemm_wmma<false><<<dim3(DMODEL / 64, N_NODES / 128), 256, 0, stream>>>(resh, woh, bo, d_out, HID, DMODEL);
}
